// Quantize_65412351918207
// MI455X (gfx1250) — matrix operand order repaired and checked by reading, not compiled, not run on hardware
//
#include <hip/hip_runtime.h>
#include <cstdint>

#define DIM 256
#define NEMB 2048
#define NROWS 65536
#define DECAY 0.99f
#define OMD 0.01f
#define EPSV 1e-5f

typedef __attribute__((ext_vector_type(16))) __bf16 v16bf;
typedef __attribute__((ext_vector_type(8)))  float  v8f;
typedef int vint4 __attribute__((vector_size(16)));

union BfPack { __bf16 h[2]; uint32_t u; };

// ---- CDNA5 async global->LDS copy (ASYNCcnt path), with guarded fallback
#if defined(__has_builtin)
# if __has_builtin(__builtin_amdgcn_global_load_async_to_lds_b128)
#  define HAVE_ASYNC_LDS 1
# endif
# if __has_builtin(__builtin_amdgcn_s_wait_asynccnt)
#  define HAVE_WAIT_ASYNC_BUILTIN 1
# endif
#endif
#ifndef HAVE_ASYNC_LDS
# define HAVE_ASYNC_LDS 0
#endif
#ifndef HAVE_WAIT_ASYNC_BUILTIN
# define HAVE_WAIT_ASYNC_BUILTIN 0
#endif

#if HAVE_ASYNC_LDS
// Copy 32B per lane: two b128 async transfers (same inst_offset applies to
// both the global and LDS address per ISA 08_async_tensor). Builtin takes
// typed vector pointers: (vint4 AS1* src, vint4 AS3* dst, imm off, imm cpol).
#define ASYNC_CP32(gsrc, ldst)                                               \
  do {                                                                       \
    __builtin_amdgcn_global_load_async_to_lds_b128(                          \
        (__attribute__((address_space(1))) vint4*)(gsrc),                    \
        (__attribute__((address_space(3))) vint4*)(ldst), 0, 0);             \
    __builtin_amdgcn_global_load_async_to_lds_b128(                          \
        (__attribute__((address_space(1))) vint4*)(gsrc),                    \
        (__attribute__((address_space(3))) vint4*)(ldst), 16, 0);            \
  } while (0)
#else
#define ASYNC_CP32(gsrc, ldst)                                               \
  do {                                                                       \
    *(float4*)((char*)(ldst)) = *(const float4*)((const char*)(gsrc));       \
    *(float4*)((char*)(ldst) + 16) =                                         \
        *(const float4*)((const char*)(gsrc) + 16);                          \
  } while (0)
#endif

__device__ __forceinline__ void wait_async0() {
#if HAVE_ASYNC_LDS
# if HAVE_WAIT_ASYNC_BUILTIN
  __builtin_amdgcn_s_wait_asynccnt(0);
# else
  asm volatile("s_wait_asynccnt 0" ::: "memory");
# endif
#endif
}

// ---------------------------------------------------------------- zero init
__global__ __launch_bounds__(256) void zero_ws_kernel(float* a, int na,
                                                      float* b, int nb) {
  int idx = blockIdx.x * blockDim.x + threadIdx.x;
  int stride = gridDim.x * blockDim.x;
  for (int i = idx; i < na; i += stride) a[i] = 0.0f;
  for (int i = idx; i < nb; i += stride) b[i] = 0.0f;
}

// ---------------------------------------------------------------- ||e_j||^2
__global__ __launch_bounds__(256) void colnorm_kernel(
    const float* __restrict__ embed, float* __restrict__ colnorm) {
  int col = blockIdx.x * blockDim.x + threadIdx.x;
  if (col >= NEMB) return;
  float s = 0.f;
  for (int k = 0; k < DIM; ++k) {
    float e = embed[(size_t)k * NEMB + col];
    s += e * e;
  }
  colnorm[col] = s;
}

// ------------------------------------------- pack embed -> bf16 B fragments
// Storage: frag(tn, kc) = embedB[((tn*8+kc)*32 + lane)*8 + v]
// ISA B layout (16-bit, 32x16): lane -> col = lane%16, K = 32*kc + 16*(v>>2) + 2*(v&3) + 8*(lane/16)
__global__ __launch_bounds__(256) void pack_embed_kernel(
    const float* __restrict__ embed, uint32_t* __restrict__ embedB) {
  int idx = blockIdx.x * blockDim.x + threadIdx.x;  // 262144 total
  int v    = idx & 7;
  int lane = (idx >> 3) & 31;
  int kc   = (idx >> 8) & 7;
  int tn   = idx >> 11;
  int col  = tn * 16 + (lane & 15);
  int hi   = lane >> 4;
  int K    = 32 * kc + 16 * (v >> 2) + 2 * (v & 3) + 8 * hi;
  BfPack p;
  p.h[0] = (__bf16)embed[(size_t)K * NEMB + col];
  p.h[1] = (__bf16)embed[(size_t)(K + 1) * NEMB + col];
  embedB[idx] = p.u;
}

// --------------------------------------------------- fused GEMM + argmin
// One wave per 16-row tile; 8 waves per block share B tiles staged in LDS by
// double-buffered async global->LDS copies. dist_cmp(n) = ||e_n||^2 - 2 x.e_n
// (||x||^2 is row-constant and cannot change the argmin, so it is dropped).
__global__ __launch_bounds__(256) void argmin_wmma_kernel(
    const float* __restrict__ flat, const uint32_t* __restrict__ embedB,
    const float* __restrict__ colnorm, int* __restrict__ ind_i32,
    float* __restrict__ ind_out) {
  __shared__ float s_coln[NEMB];           // 8 KB
  __shared__ uint32_t s_bfrag[2][2048];    // 2 x 8 KB double buffer
  for (int j = threadIdx.x; j < NEMB; j += blockDim.x) s_coln[j] = colnorm[j];

  int lane = threadIdx.x & 31;
  int waveInBlock = threadIdx.x >> 5;      // == kc chunk this wave stages
  int rowTile = blockIdx.x * 8 + waveInBlock;  // 4096 tiles total
  int hi = lane >> 4;
  int lo = lane & 15;

  // A fragments: ISA 16-bit A layout 16x32 per kc chunk.
  // lane -> row = lane%16; dword v -> K = 32*kc + 16*(v>>2) + 2*(v&3) + 8*hi
  const float* xrow = flat + (size_t)(rowTile * 16 + lo) * DIM;
  v16bf a[8];
#pragma unroll
  for (int kc = 0; kc < 8; ++kc) {
    v16bf af;
#pragma unroll
    for (int v = 0; v < 8; ++v) {
      int K = 32 * kc + 16 * (v >> 2) + 2 * (v & 3) + 8 * hi;
      float2 f = *reinterpret_cast<const float2*>(xrow + K);
      af[2 * v]     = (__bf16)f.x;
      af[2 * v + 1] = (__bf16)f.y;
    }
    a[kc] = af;
  }

  float best[8];
  int   bidx[8];
#pragma unroll
  for (int v = 0; v < 8; ++v) { best[v] = 3.0e38f; bidx[v] = 0; }

  // Prologue: stage tile 0 (wave w stages its kc = w chunk: 32B per lane).
  {
    const uint32_t* g = embedB + ((size_t)(0 * 8 + waveInBlock) * 32 + lane) * 8;
    uint32_t* l = &s_bfrag[0][(waveInBlock * 32 + lane) * 8];
    ASYNC_CP32(g, l);
  }

  for (int tn = 0; tn < 128; ++tn) {
    int cur = tn & 1;
    int nxt = cur ^ 1;
    wait_async0();        // my async chunk of tile tn is resident
    __syncthreads();      // everyone's chunk resident; prior reads of buf[nxt]
                          // were consumed last iteration (dscnt waited)
    if (tn + 1 < 128) {
      const uint32_t* g =
          embedB + ((size_t)((tn + 1) * 8 + waveInBlock) * 32 + lane) * 8;
      uint32_t* l = &s_bfrag[nxt][(waveInBlock * 32 + lane) * 8];
      ASYNC_CP32(g, l);
    }
    if (tn + 2 < 128)
      __builtin_prefetch(embedB + ((size_t)(tn + 2) * 8 * 32 + lane) * 8, 0, 0);

    // Two independent accumulator chains over K halves to shorten the
    // WMMA->WMMA dependency chain; dist only needs c0+c1.
    v8f c0 = {};
    v8f c1 = {};
#pragma unroll
    for (int kc = 0; kc < 4; ++kc) {
      v16bf b = *reinterpret_cast<const v16bf*>(&s_bfrag[cur][(kc * 32 + lane) * 8]);
      c0 = __builtin_amdgcn_wmma_f32_16x16x32_bf16(false, a[kc], false, b,
                                                   (short)0, c0, false, false);
    }
#pragma unroll
    for (int kc = 4; kc < 8; ++kc) {
      v16bf b = *reinterpret_cast<const v16bf*>(&s_bfrag[cur][(kc * 32 + lane) * 8]);
      c1 = __builtin_amdgcn_wmma_f32_16x16x32_bf16(false, a[kc], false, b,
                                                   (short)0, c1, false, false);
    }

    // C layout: vgpr v, lanes 0-15 -> (M=v, N=lane); lanes 16-31 -> (M=v+8)
    int coln = tn * 16 + lo;
    float cn = s_coln[coln];
#pragma unroll
    for (int v = 0; v < 8; ++v) {
      float d = cn - 2.0f * (c0[v] + c1[v]);
      if (d < best[v]) { best[v] = d; bidx[v] = coln; }
    }
  }

  // Min-reduce over the 16 lanes that share each row set (xor < 16 never
  // crosses the lane-16 boundary). Tie-break: lowest index (matches argmin).
#pragma unroll
  for (int off = 1; off < 16; off <<= 1) {
#pragma unroll
    for (int v = 0; v < 8; ++v) {
      float ob = __shfl_xor(best[v], off, 32);
      int   oi = __shfl_xor(bidx[v], off, 32);
      if (ob < best[v] || (ob == best[v] && oi < bidx[v])) {
        best[v] = ob; bidx[v] = oi;
      }
    }
  }

  if (lo == 0) {
    int rowBase = rowTile * 16 + 8 * hi;
#pragma unroll
    for (int v = 0; v < 8; ++v) {
      ind_i32[rowBase + v] = bidx[v];
      ind_out[rowBase + v] = (float)bidx[v];
    }
  }
}

// ----------------------------------- gather codes, diff, EMA segment sums
__global__ __launch_bounds__(256) void scatter_kernel(
    const float* __restrict__ flat, const float* __restrict__ embed,
    const int* __restrict__ ind_i32, float* __restrict__ qout,
    float* __restrict__ embed_sum, float* __restrict__ cnt,
    float* __restrict__ diff_acc) {
  int row = blockIdx.x;
  int k = threadIdx.x;
  int ci = ind_i32[row];
  float e = embed[(size_t)k * NEMB + ci];   // embed is L2-resident (2 MB)
  float x = flat[(size_t)row * DIM + k];
  qout[(size_t)row * DIM + k] = e;          // quantize_st == quantize value

  float d = (e - x) * (e - x);
  __shared__ float sred[256];
  sred[k] = d;
  __syncthreads();
  for (int s = 128; s > 0; s >>= 1) {
    if (k < s) sred[k] += sred[k + s];
    __syncthreads();
  }
  if (k == 0) {
    atomicAdd(diff_acc, sred[0]);
    atomicAdd(&cnt[ci], 1.0f);
  }
  atomicAdd(&embed_sum[(size_t)k * NEMB + ci], x);
}

// ---------------------------------------- cluster-size EMA + smoothing + diff
__global__ __launch_bounds__(256) void finalize_small_kernel(
    const float* __restrict__ cluster_size, const float* __restrict__ cnt,
    float* __restrict__ smoothed, float* __restrict__ out_ncs,
    const float* __restrict__ diff_acc, float* __restrict__ out_diff) {
  __shared__ float sred[256];
  int tid = threadIdx.x;
  float ncsv[8];
  float local = 0.f;
#pragma unroll
  for (int j8 = 0; j8 < 8; ++j8) {
    int j = tid + 256 * j8;
    float v = cluster_size[j] * DECAY + OMD * cnt[j];
    ncsv[j8] = v;
    out_ncs[j] = v;
    local += v;
  }
  sred[tid] = local;
  __syncthreads();
  for (int s = 128; s > 0; s >>= 1) {
    if (tid < s) sred[tid] += sred[tid + s];
    __syncthreads();
  }
  float n = sred[0];
  float scale = n / (n + (float)NEMB * EPSV);
#pragma unroll
  for (int j8 = 0; j8 < 8; ++j8) {
    int j = tid + 256 * j8;
    smoothed[j] = (ncsv[j8] + EPSV) * scale;
  }
  if (tid == 0) out_diff[0] = diff_acc[0] * (1.0f / 16777216.0f);
}

// ------------------------------------------- embed_avg EMA and new codebook
__global__ __launch_bounds__(256) void finalize_embed_kernel(
    const float* __restrict__ embed_avg, const float* __restrict__ embed_sum,
    const float* __restrict__ smoothed, float* __restrict__ out_embed_avg,
    float* __restrict__ out_embed) {
  int idx = blockIdx.x * blockDim.x + threadIdx.x;  // 524288
  int col = idx & (NEMB - 1);
  float ea = embed_avg[idx] * DECAY + OMD * embed_sum[idx];
  out_embed_avg[idx] = ea;
  out_embed[idx] = ea / smoothed[col];
}

// ---------------------------------------------------------------- launcher
extern "C" void kernel_launch(void* const* d_in, const int* in_sizes, int n_in,
                              void* d_out, int out_size, void* d_ws,
                              size_t ws_size, hipStream_t stream) {
  const float* inputs       = (const float*)d_in[0];  // [16,64,64,256]
  const float* embed        = (const float*)d_in[1];  // [256,2048]
  const float* cluster_size = (const float*)d_in[2];  // [2048]
  const float* embed_avg    = (const float*)d_in[3];  // [256,2048]
  float* out = (float*)d_out;

  // Output layout (flat floats, return order)
  const size_t Q_OFF    = 0;                   // 16777216 quantize_st
  const size_t DIFF_OFF = 16777216;            // 1        diff
  const size_t IND_OFF  = 16777217;            // 65536    embed_ind
  const size_t NE_OFF   = 16842753;            // 524288   new_embed
  const size_t NCS_OFF  = 17367041;            // 2048     new_cluster_size
  const size_t NEA_OFF  = 17369089;            // 524288   new_embed_avg

  // Workspace layout (bytes)
  char* ws = (char*)d_ws;
  int*      ind_i32  = (int*)     (ws + 0);        // 262144 B
  float*    colnorm  = (float*)   (ws + 262144);   // 8192 B
  uint32_t* embedB   = (uint32_t*)(ws + 270336);   // 1048576 B (bf16 B frags)
  float*    cnt      = (float*)   (ws + 1318912);  // 8192 B
  float*    embed_sum= (float*)   (ws + 1327104);  // 2097152 B
  float*    smoothed = (float*)   (ws + 3424256);  // 8192 B
  float*    diff_acc = (float*)   (ws + 3432448);  // 4 B

  // cnt + embed_sum are contiguous: zero both in one pass, plus diff_acc.
  zero_ws_kernel<<<256, 256, 0, stream>>>(cnt, NEMB + DIM * NEMB, diff_acc, 1);
  colnorm_kernel<<<NEMB / 256, 256, 0, stream>>>(embed, colnorm);
  pack_embed_kernel<<<1024, 256, 0, stream>>>(embed, embedB);

  // 4096 row-tiles, 8 waves/block -> 512 blocks
  argmin_wmma_kernel<<<512, 256, 0, stream>>>(inputs, embedB, colnorm,
                                              ind_i32, out + IND_OFF);

  scatter_kernel<<<NROWS, 256, 0, stream>>>(inputs, embed, ind_i32,
                                            out + Q_OFF, embed_sum, cnt,
                                            diff_acc);

  finalize_small_kernel<<<1, 256, 0, stream>>>(cluster_size, cnt, smoothed,
                                               out + NCS_OFF, diff_acc,
                                               out + DIFF_OFF);

  finalize_embed_kernel<<<(DIM * NEMB) / 256, 256, 0, stream>>>(
      embed_avg, embed_sum, smoothed, out + NEA_OFF, out + NE_OFF);
}
